// ALLGAT_61125974557022
// MI455X (gfx1250) — compile-verified, matrix operand order checked by reading, not run on hardware
//
#include <hip/hip_runtime.h>

// ---------------- types for WMMA ----------------
typedef __attribute__((ext_vector_type(16))) __bf16 v16bf;
typedef __attribute__((ext_vector_type(8)))  float  v8f;
typedef __attribute__((ext_vector_type(4)))  unsigned int u32x4;

union FragU { u32x4 q[2]; v16bf v; };

__device__ __forceinline__ v16bf load_frag16(const unsigned short* p) {
    // 16 bf16 = two 16-byte loads: K = kb+0..7 at p[0..7], K = kb+16..23 at p[16..23]
    FragU f;
    f.q[0] = *(const u32x4*)(p);
    f.q[1] = *(const u32x4*)(p + 16);
    return f.v;
}

__device__ __forceinline__ v8f wmma_bf16(v16bf a, v16bf b, v8f c) {
    return __builtin_amdgcn_wmma_f32_16x16x32_bf16(
        /*neg_a=*/false, a, /*neg_b=*/false, b,
        /*c_mod=*/(short)0, c, /*reuse_a=*/false, /*reuse_b=*/false);
}

__device__ __forceinline__ unsigned short f2bf(float f) {
    unsigned int u = __float_as_uint(f);
    unsigned int r = u + 0x7FFFu + ((u >> 16) & 1u);   // round-to-nearest-even
    return (unsigned short)(r >> 16);
}

__device__ __forceinline__ void atomicMaxFloat(float* addr, float value) {
    // exact float max via signed/unsigned int ordering trick
    if (value >= 0.f) atomicMax((int*)addr, __float_as_int(value));
    else              atomicMin((unsigned int*)addr, __float_as_uint(value));
}

#define GN 50000
#define GE 800000
#define GD 128
#define GH 8
#define GDFF 512

// ---------------- 1: repack x -> bf16, weights -> bf16 column-major ----------------
__global__ void repack_kernel(const float* __restrict__ x, const float* __restrict__ Wfc,
                              const float* __restrict__ W1, const float* __restrict__ W2,
                              unsigned short* __restrict__ xb, unsigned short* __restrict__ Wzt,
                              unsigned short* __restrict__ W1t, unsigned short* __restrict__ W2t) {
    int idx = blockIdx.x * blockDim.x + threadIdx.x;
    if (idx < GN * GD) xb[idx] = f2bf(x[idx]);
    if (idx < GD * GD) {                    // Wzt[j*128+d] = Wfc[h,d,k], j = h*16+k
        int j = idx >> 7, d = idx & 127;
        int h = j >> 4, k = j & 15;
        Wzt[idx] = f2bf(Wfc[(h * GD + d) * 16 + k]);
    }
    if (idx < GDFF * GD) {
        int f = idx >> 7, d = idx & 127;    // W1t[f*128+d] = W1[d*512+f]
        W1t[idx] = f2bf(W1[d * GDFF + f]);
        int d2 = idx >> 9, f2 = idx & 511;  // W2t[d2*512+f2] = W2[f2*128+d2]
        W2t[idx] = f2bf(W2[f2 * GD + d2]);
    }
}

// ---------------- 2: init accumulators ----------------
__global__ void init_kernel(float* __restrict__ hagg, float* __restrict__ m, float* __restrict__ s) {
    int idx = blockIdx.x * blockDim.x + threadIdx.x;
    if (idx < GN * GD) hagg[idx] = 0.f;
    if (idx < GN * GH) { m[idx] = -INFINITY; s[idx] = 0.f; }
}

// ---------------- 3: z = x @ Wz  (bf16 WMMA, K=128) ----------------
__global__ void gemm_z_kernel(const unsigned short* __restrict__ xb,
                              const unsigned short* __restrict__ Wzt,
                              float* __restrict__ z) {
    int wave = threadIdx.x >> 5, lane = threadIdx.x & 31;
    int rowBase = blockIdx.x * 16;
    int col0 = wave * 16;                 // 8 waves cover 128 output cols
    int lrow = lane & 15;
    int kb = (lane & 16) ? 8 : 0;         // ISA 16-bit A/B lane split
    const unsigned short* aRow = xb  + (size_t)(rowBase + lrow) * GD + kb;
    const unsigned short* bCol = Wzt + (size_t)(col0 + lrow)   * GD + kb;
    v8f acc = {};
    for (int k0 = 0; k0 < GD; k0 += 32)
        acc = wmma_bf16(load_frag16(aRow + k0), load_frag16(bCol + k0), acc);
    int half8 = (lane & 16) >> 1;         // lanes 16-31 hold rows M=8..15
    int n = col0 + lrow;
    for (int j = 0; j < 8; ++j)
        z[(size_t)(rowBase + half8 + j) * GD + n] = acc[j];
}

// ---------------- 4: el/er = per-head dot of z with a_l/a_r ----------------
__global__ void elr_kernel(const float* __restrict__ z, const float* __restrict__ a_l,
                           const float* __restrict__ a_r, float* __restrict__ el,
                           float* __restrict__ er) {
    int idx = blockIdx.x * blockDim.x + threadIdx.x;
    if (idx >= GN * GH) return;
    int h = idx & 7;
    const float* zp = z + (size_t)(idx >> 3) * GD + h * 16;
    float sl = 0.f, sr = 0.f;
    #pragma unroll
    for (int k = 0; k < 16; ++k) { sl += zp[k] * a_l[h * 16 + k]; sr += zp[k] * a_r[h * 16 + k]; }
    el[idx] = sl; er[idx] = sr;
}

// ---------------- 5: edge logits + segment max ----------------
__global__ void edge1_kernel(const float* __restrict__ el, const float* __restrict__ er,
                             const int* __restrict__ esrc, const int* __restrict__ edst,
                             float* __restrict__ ebuf, float* __restrict__ m) {
    int idx = blockIdx.x * blockDim.x + threadIdx.x;
    if (idx >= GE * GH) return;
    int e = idx >> 3, h = idx & 7;
    float v = el[esrc[e] * GH + h] + er[edst[e] * GH + h];
    v = v > 0.f ? v : 0.01f * v;          // leaky_relu
    ebuf[idx] = v;
    atomicMaxFloat(&m[edst[e] * GH + h], v);
}

// ---------------- 6: exp(e - m[dst]) + segment sum ----------------
__global__ void edge2_kernel(const int* __restrict__ edst, const float* __restrict__ m,
                             float* __restrict__ ebuf, float* __restrict__ s) {
    int idx = blockIdx.x * blockDim.x + threadIdx.x;
    if (idx >= GE * GH) return;
    int e = idx >> 3, h = idx & 7;
    float mm = m[edst[e] * GH + h];
    if (mm < -3.0e38f) mm = 0.f;          // empty-segment guard (m == -inf)
    float ee = __expf(ebuf[idx] - mm);
    ebuf[idx] = ee;
    atomicAdd(&s[edst[e] * GH + h], ee);
}

// ---------------- 7: weighted scatter-aggregate (memory-bound hot loop) ----------------
__global__ void scatter_kernel(const float* __restrict__ z, const float* __restrict__ ebuf,
                               const float* __restrict__ s, const int* __restrict__ esrc,
                               const int* __restrict__ edst, float* __restrict__ hagg) {
    int e = blockIdx.x * 2 + (threadIdx.x >> 7);  // 2 edges per 256-thread block
    int t = threadIdx.x & 127;                    // feature index: h = t/16, k = t%16
    int src = esrc[e], dst = edst[e];
    int h = t >> 4;
    float alpha = ebuf[(size_t)e * GH + h] / fmaxf(s[dst * GH + h], 1e-9f);
    atomicAdd(&hagg[(size_t)dst * GD + t], alpha * z[(size_t)src * GD + t]);
}

// ---------------- 8: ELU + residual + LayerNorm (one row per 128-thread block) ----------------
__global__ void node_ln_kernel(const float* __restrict__ hagg, const float* __restrict__ x,
                               const float* __restrict__ gamma, const float* __restrict__ beta,
                               float* __restrict__ hres, unsigned short* __restrict__ lnb) {
    int row = blockIdx.x;
    int t = threadIdx.x;
    float v = hagg[(size_t)row * GD + t];
    v = v > 0.f ? v : (__expf(v) - 1.f);          // elu
    v += x[(size_t)row * GD + t];                 // residual
    hres[(size_t)row * GD + t] = v;
    __shared__ float rs[GD];
    __shared__ float rq[GD];
    rs[t] = v; rq[t] = v * v;
    __syncthreads();
    for (int sft = 64; sft > 0; sft >>= 1) {
        if (t < sft) { rs[t] += rs[t + sft]; rq[t] += rq[t + sft]; }
        __syncthreads();
    }
    float mean = rs[0] * (1.f / GD);
    float var  = rq[0] * (1.f / GD) - mean * mean;
    float ln = (v - mean) * rsqrtf(var + 1e-5f) * gamma[t] + beta[t];
    lnb[(size_t)row * GD + t] = f2bf(ln);
}

// ---------------- 9: inter = relu(ln @ W1 + b1)  (bf16 WMMA, K=128, 512 cols) ----------------
__global__ void gemm_ffn1_kernel(const unsigned short* __restrict__ lnb,
                                 const unsigned short* __restrict__ W1t,
                                 const float* __restrict__ b1,
                                 unsigned short* __restrict__ inter) {
    int wave = threadIdx.x >> 5, lane = threadIdx.x & 31;
    int rowBase = blockIdx.x * 16;
    int col0 = (blockIdx.y * 8 + wave) * 16;      // 4 * 8 * 16 = 512 cols
    int lrow = lane & 15;
    int kb = (lane & 16) ? 8 : 0;
    const unsigned short* aRow = lnb + (size_t)(rowBase + lrow) * GD + kb;
    const unsigned short* bCol = W1t + (size_t)(col0 + lrow)   * GD + kb;
    v8f acc = {};
    for (int k0 = 0; k0 < GD; k0 += 32)
        acc = wmma_bf16(load_frag16(aRow + k0), load_frag16(bCol + k0), acc);
    int half8 = (lane & 16) >> 1;
    int n = col0 + lrow;
    float bias = b1[n];
    for (int j = 0; j < 8; ++j) {
        float v = fmaxf(acc[j] + bias, 0.f);      // relu epilogue
        inter[(size_t)(rowBase + half8 + j) * GDFF + n] = f2bf(v);
    }
}

// ---------------- 10: out = inter @ W2 + b2 + hres  (bf16 WMMA, K=512) ----------------
__global__ void gemm_ffn2_kernel(const unsigned short* __restrict__ inter,
                                 const unsigned short* __restrict__ W2t,
                                 const float* __restrict__ b2,
                                 const float* __restrict__ hres,
                                 float* __restrict__ out) {
    int wave = threadIdx.x >> 5, lane = threadIdx.x & 31;
    int rowBase = blockIdx.x * 16;
    int col0 = wave * 16;
    int lrow = lane & 15;
    int kb = (lane & 16) ? 8 : 0;
    const unsigned short* aRow = inter + (size_t)(rowBase + lrow) * GDFF + kb;
    const unsigned short* bCol = W2t  + (size_t)(col0 + lrow)    * GDFF + kb;
    v8f acc = {};
    for (int k0 = 0; k0 < GDFF; k0 += 32)
        acc = wmma_bf16(load_frag16(aRow + k0), load_frag16(bCol + k0), acc);
    int half8 = (lane & 16) >> 1;
    int n = col0 + lrow;
    float bias = b2[n];
    for (int j = 0; j < 8; ++j) {
        size_t r = (size_t)(rowBase + half8 + j);
        out[r * GD + n] = acc[j] + bias + hres[r * GD + n];
    }
}

// ---------------- launch ----------------
extern "C" void kernel_launch(void* const* d_in, const int* in_sizes, int n_in,
                              void* d_out, int out_size, void* d_ws, size_t ws_size,
                              hipStream_t stream) {
    const float* x     = (const float*)d_in[0];
    const float* Wfc   = (const float*)d_in[1];
    const float* a_l   = (const float*)d_in[2];
    const float* a_r   = (const float*)d_in[3];
    const float* gamma = (const float*)d_in[4];
    const float* beta  = (const float*)d_in[5];
    const float* W1    = (const float*)d_in[6];
    const float* b1    = (const float*)d_in[7];
    const float* W2    = (const float*)d_in[8];
    const float* b2    = (const float*)d_in[9];
    const int* esrc    = (const int*)d_in[10];
    const int* edst    = (const int*)d_in[11];
    float* out = (float*)d_out;

    char* ws = (char*)d_ws;
    size_t off = 0;
    auto take = [&](size_t bytes) -> char* {
        char* p = ws + off;
        off += (bytes + 255) & ~(size_t)255;
        return p;
    };
    unsigned short* xb    = (unsigned short*)take((size_t)GN * GD * 2);
    unsigned short* Wzt   = (unsigned short*)take((size_t)GD * GD * 2);
    unsigned short* W1t   = (unsigned short*)take((size_t)GDFF * GD * 2);
    unsigned short* W2t   = (unsigned short*)take((size_t)GD * GDFF * 2);
    float*          z     = (float*)take((size_t)GN * GD * 4);
    float*          el    = (float*)take((size_t)GN * GH * 4);
    float*          er    = (float*)take((size_t)GN * GH * 4);
    float*          m     = (float*)take((size_t)GN * GH * 4);
    float*          s     = (float*)take((size_t)GN * GH * 4);
    float*          ebuf  = (float*)take((size_t)GE * GH * 4);
    float*          hagg  = (float*)take((size_t)GN * GD * 4);
    float*          hres  = (float*)take((size_t)GN * GD * 4);
    unsigned short* lnb   = (unsigned short*)take((size_t)GN * GD * 2);
    unsigned short* inter = (unsigned short*)take((size_t)GN * GDFF * 2);

    const int mtiles = GN / 16;  // 3125, exact

    repack_kernel<<<(GN * GD + 255) / 256, 256, 0, stream>>>(x, Wfc, W1, W2, xb, Wzt, W1t, W2t);
    init_kernel  <<<(GN * GD + 255) / 256, 256, 0, stream>>>(hagg, m, s);
    gemm_z_kernel<<<mtiles, 256, 0, stream>>>(xb, Wzt, z);
    elr_kernel   <<<(GN * GH + 255) / 256, 256, 0, stream>>>(z, a_l, a_r, el, er);
    edge1_kernel <<<(GE * GH + 255) / 256, 256, 0, stream>>>(el, er, esrc, edst, ebuf, m);
    edge2_kernel <<<(GE * GH + 255) / 256, 256, 0, stream>>>(edst, m, ebuf, s);
    scatter_kernel<<<GE / 2, 256, 0, stream>>>(z, ebuf, s, esrc, edst, hagg);
    node_ln_kernel<<<GN, GD, 0, stream>>>(hagg, x, gamma, beta, hres, lnb);
    gemm_ffn1_kernel<<<dim3(mtiles, 4), 256, 0, stream>>>(lnb, W1t, b1, inter);
    gemm_ffn2_kernel<<<mtiles, 256, 0, stream>>>(inter, W2t, b2, hres, out);
}
